// convex_hull_cvx_dyn_66795331387675
// MI455X (gfx1250) — matrix operand order repaired and checked by reading, not compiled, not run on hardware
//
#include <hip/hip_runtime.h>
#include <hip/hip_bf16.h>
#include <cmath>

// ---------------------------------------------------------------------------
// Problem constants (from the reference)
// ---------------------------------------------------------------------------
#define NROWS 2048
#define DCOLS 512
#define N_ITERS 200
#define BISECT_ITERS 50
#define POWER_ITERS 30

typedef __attribute__((ext_vector_type(16))) __bf16 v16bf;
typedef __attribute__((ext_vector_type(8)))  __bf16 v8bf;
typedef __attribute__((ext_vector_type(8)))  float  v8f;

union Frag16 { v16bf v; v8bf h[2]; };

// ---------------------------------------------------------------------------
// Row normalize: out_f32 = row / ||row||, plus bf16 copy for WMMA operands
// ---------------------------------------------------------------------------
__global__ __launch_bounds__(256) void row_normalize(
    const float* __restrict__ in, float* __restrict__ outf,
    __bf16* __restrict__ outh, int D)
{
    __shared__ float red[256];
    const int row = blockIdx.x, tid = threadIdx.x;
    const float* rp = in + (size_t)row * D;
    float s = 0.f;
    for (int j = tid; j < D; j += 256) { float t = rp[j]; s += t * t; }
    red[tid] = s; __syncthreads();
    for (int o = 128; o; o >>= 1) { if (tid < o) red[tid] += red[tid + o]; __syncthreads(); }
    const float inv = 1.0f / sqrtf(red[0]);
    for (int j = tid; j < D; j += 256) {
        float t = rp[j] * inv;
        outf[(size_t)row * D + j] = t;
        outh[(size_t)row * D + j] = (__bf16)t;
    }
}

// ---------------------------------------------------------------------------
// WMMA GEMM, Out[i][j] = sum_k Ah[i][k] * Bh[j][k]  (A·B^T, both row-major).
// Q is symmetric so Z·Q is also this form with B=Qh.
// Block tile 128x128, K tile 32. 8 waves (wave32): 4x2 wave grid, each wave
// owns a 32x64 sub-tile = 2x4 WMMA 16x16 accumulators.
// MODE 0: Out = acc (f32)            -- used for Q = A2·A2^T and C = A1·A2^T
// MODE 1: Out = Z - 2*step*(acc - C) -- fused FISTA gradient step (V matrix)
// ---------------------------------------------------------------------------
template <int MODE>
__global__ __launch_bounds__(256) void wmma_gemm_abt(
    const __bf16* __restrict__ Ah, const __bf16* __restrict__ Bh,
    int M, int Nn, int K,
    float* __restrict__ Out,
    const float* __restrict__ Zf, const float* __restrict__ Cf,
    const float* __restrict__ stepPtr)
{
    constexpr int BM = 128, BN = 128, BK = 32;
    constexpr int SA = BK + 8, SB = BK + 8;        // padded LDS strides (bf16 units)
    __shared__ __bf16 ldsA[BM * SA];
    __shared__ __bf16 ldsB[BN * SB];

    const int tid   = threadIdx.x;
    const int wave  = tid >> 5;
    const int lane  = tid & 31;
    const int waveM = wave >> 1;                    // 0..3  -> 32-row slabs
    const int waveN = wave & 1;                     // 0..1  -> 64-col slabs
    const int r16   = lane & 15;
    const int hi    = lane >> 4;                    // half-wave select
    const int mBase = blockIdx.y * BM;
    const int nBase = blockIdx.x * BN;

    v8f acc[2][4] = {};

    for (int kt = 0; kt < K; kt += BK) {
        __syncthreads();
        // Stage A and B tiles (rows of A / rows of B are both contiguous in K).
#pragma unroll
        for (int c = 0; c < 2; ++c) {
            int id   = tid + c * 256;               // 0..511
            int row  = id >> 2;                     // 0..127
            int col8 = (id & 3) * 8;                // 0,8,16,24
            *(uint4*)&ldsA[row * SA + col8] =
                *(const uint4*)&Ah[(size_t)(mBase + row) * K + kt + col8];
            *(uint4*)&ldsB[row * SB + col8] =
                *(const uint4*)&Bh[(size_t)(nBase + row) * K + kt + col8];
        }
        __syncthreads();

        // A fragment (16x32 bf16): lanes 0-15 rows M, K = 0..7 & 16..23;
        // lanes 16-31 same rows, K = 8..15 & 24..31.
        Frag16 a[2], b[4];
        const int kbA = hi * 8;
#pragma unroll
        for (int i = 0; i < 2; ++i) {
            int rowA = waveM * 32 + i * 16 + r16;
            a[i].h[0] = *(const v8bf*)&ldsA[rowA * SA + kbA];
            a[i].h[1] = *(const v8bf*)&ldsA[rowA * SA + kbA + 16];
        }
        // B fragment (32x16 bf16): lane holds column N=lane%16, same K order as A:
        // lanes 0-15 K = 0..7 & 16..23; lanes 16-31 K = 8..15 & 24..31.
        const int kbB = hi * 8;
#pragma unroll
        for (int j = 0; j < 4; ++j) {
            int rowB = waveN * 64 + j * 16 + r16;
            b[j].h[0] = *(const v8bf*)&ldsB[rowB * SB + kbB];
            b[j].h[1] = *(const v8bf*)&ldsB[rowB * SB + kbB + 16];
        }
#pragma unroll
        for (int i = 0; i < 2; ++i)
#pragma unroll
            for (int j = 0; j < 4; ++j)
                acc[i][j] = __builtin_amdgcn_wmma_f32_16x16x32_bf16(
                    false, a[i].v, false, b[j].v, (short)0, acc[i][j], false, false);
    }

    const float step2 = (MODE == 1) ? 2.0f * stepPtr[0] : 0.0f;
#pragma unroll
    for (int i = 0; i < 2; ++i)
#pragma unroll
        for (int j = 0; j < 4; ++j) {
            const int gj = nBase + waveN * 64 + j * 16 + r16;
#pragma unroll
            for (int e = 0; e < 8; ++e) {
                const int gi  = mBase + waveM * 32 + i * 16 + hi * 8 + e;
                const size_t idx = (size_t)gi * Nn + gj;
                float v = acc[i][j][e];
                if (MODE == 1) v = Zf[idx] - step2 * (v - Cf[idx]);
                Out[idx] = v;
            }
        }
}

// ---------------------------------------------------------------------------
// Power iteration helpers (Q kept in f32; tiny cost vs the GEMMs)
// ---------------------------------------------------------------------------
__global__ __launch_bounds__(256) void matvec2048(
    const float* __restrict__ Q, const float* __restrict__ v,
    float* __restrict__ w, int N)
{
    const int wave = threadIdx.x >> 5, lane = threadIdx.x & 31;
    const int row  = blockIdx.x * 8 + wave;
    const float* qr = Q + (size_t)row * N;
    float s = 0.f;
    for (int j = lane; j < N; j += 32) s += qr[j] * v[j];
    for (int off = 16; off; off >>= 1) s += __shfl_xor(s, off, 32);
    if (lane == 0) w[row] = s;
}

__global__ __launch_bounds__(256) void vec_normalize(
    const float* __restrict__ w, float* __restrict__ v, int N)
{
    __shared__ float red[256];
    const int tid = threadIdx.x;
    float s = 0.f;
    for (int j = tid; j < N; j += 256) s += w[j] * w[j];
    red[tid] = s; __syncthreads();
    for (int o = 128; o; o >>= 1) { if (tid < o) red[tid] += red[tid + o]; __syncthreads(); }
    const float inv = 1.0f / (sqrtf(red[0]) + 1e-12f);
    for (int j = tid; j < N; j += 256) v[j] = w[j] * inv;
}

__global__ __launch_bounds__(256) void dot_step(
    const float* __restrict__ v, const float* __restrict__ w,
    float* __restrict__ scal, int N)
{
    __shared__ float red[256];
    const int tid = threadIdx.x;
    float s = 0.f;
    for (int j = tid; j < N; j += 256) s += v[j] * w[j];
    red[tid] = s; __syncthreads();
    for (int o = 128; o; o >>= 1) { if (tid < o) red[tid] += red[tid + o]; __syncthreads(); }
    if (tid == 0) scal[0] = 1.0f / (2.0f * red[0] + 1e-12f);   // FISTA step size
}

// ---------------------------------------------------------------------------
// Small fills / conversions
// ---------------------------------------------------------------------------
__global__ void fill_f32(float* __restrict__ p, float val, int n) {
    int i = blockIdx.x * 256 + threadIdx.x;
    if (i < n) p[i] = val;
}
__global__ void f32_to_bf16(const float* __restrict__ in, __bf16* __restrict__ out, size_t n) {
    size_t i = (size_t)blockIdx.x * 256 + threadIdx.x;
    if (i < n) out[i] = (__bf16)in[i];
}
__global__ void init_xz(float* __restrict__ X, float* __restrict__ Z,
                        __bf16* __restrict__ Zh, float c, size_t n) {
    size_t i = (size_t)blockIdx.x * 256 + threadIdx.x;
    if (i < n) { X[i] = c; Z[i] = c; Zh[i] = (__bf16)c; }
}

// ---------------------------------------------------------------------------
// Capped-simplex projection (exact bisection, 50 steps) + FISTA momentum.
// One 256-thread block per row; row lives in registers (8 elems/thread);
// all reductions are fixed-order LDS trees -> deterministic.
// ---------------------------------------------------------------------------
__global__ __launch_bounds__(256) void fista_project(
    const float* __restrict__ V, float* __restrict__ X, float* __restrict__ Z,
    __bf16* __restrict__ Zh, int N, float mom)
{
    __shared__ float red[256];
    const int row = blockIdx.x, tid = threadIdx.x;
    const float* vr = V + (size_t)row * N;

    float x[8];
    float mn = 3.4e38f, mx = -3.4e38f;
#pragma unroll
    for (int i = 0; i < 8; ++i) {
        float t = vr[tid + 256 * i];
        x[i] = t; mn = fminf(mn, t); mx = fmaxf(mx, t);
    }
    red[tid] = mn; __syncthreads();
    for (int o = 128; o; o >>= 1) { if (tid < o) red[tid] = fminf(red[tid], red[tid + o]); __syncthreads(); }
    float lo = red[0] - 1.0f;          // min(v) - upper
    __syncthreads();
    red[tid] = mx; __syncthreads();
    for (int o = 128; o; o >>= 1) { if (tid < o) red[tid] = fmaxf(red[tid], red[tid + o]); __syncthreads(); }
    float hi = red[0];                 // max(v) - lower(=0)
    __syncthreads();

    for (int it = 0; it < BISECT_ITERS; ++it) {
        const float mid = 0.5f * (lo + hi);
        float s = 0.f;
#pragma unroll
        for (int i = 0; i < 8; ++i) s += fminf(fmaxf(x[i] - mid, 0.0f), 1.0f);
        red[tid] = s; __syncthreads();
        for (int o = 128; o; o >>= 1) { if (tid < o) red[tid] += red[tid + o]; __syncthreads(); }
        const float tot = red[0]; __syncthreads();
        if (tot > 1.0f) lo = mid; else hi = mid;   // uniform branch
    }
    const float tau = 0.5f * (lo + hi);
#pragma unroll
    for (int i = 0; i < 8; ++i) {
        const int j = tid + 256 * i;
        const size_t idx = (size_t)row * N + j;
        const float xn = fminf(fmaxf(x[i] - tau, 0.0f), 1.0f);
        const float xo = X[idx];
        const float zn = xn + mom * (xn - xo);
        X[idx]  = xn;
        Z[idx]  = zn;
        Zh[idx] = (__bf16)zn;          // bf16 operand for the next Z·Q WMMA
    }
}

// ---------------------------------------------------------------------------
// Final distance: per-block partial sums of (X@nfea2 - nfea1)^2 (f32, runs
// once, deterministic two-pass reduction).
// ---------------------------------------------------------------------------
__global__ __launch_bounds__(256) void dist_partial(
    const float* __restrict__ X, const float* __restrict__ B,
    const float* __restrict__ Y, float* __restrict__ partial, int N, int D)
{
    __shared__ float sA[16][17];
    __shared__ float sB[16][17];
    __shared__ float red[256];
    const int tx = threadIdx.x & 15, ty = threadIdx.x >> 4;
    const int rowBase = blockIdx.y * 16, colBase = blockIdx.x * 16;
    float acc = 0.f;
    for (int kt = 0; kt < N; kt += 16) {
        __syncthreads();
        sA[ty][tx] = X[(size_t)(rowBase + ty) * N + kt + tx];
        sB[ty][tx] = B[(size_t)(kt + ty) * D + colBase + tx];
        __syncthreads();
#pragma unroll
        for (int k = 0; k < 16; ++k) acc += sA[ty][k] * sB[k][tx];
    }
    const float e = acc - Y[(size_t)(rowBase + ty) * D + colBase + tx];
    red[threadIdx.x] = e * e; __syncthreads();
    for (int o = 128; o; o >>= 1) { if (threadIdx.x < o) red[threadIdx.x] += red[threadIdx.x + o]; __syncthreads(); }
    if (threadIdx.x == 0) partial[blockIdx.y * gridDim.x + blockIdx.x] = red[0];
}

__global__ __launch_bounds__(256) void final_reduce(
    const float* __restrict__ partial, float* __restrict__ out, int n, float scale)
{
    __shared__ float red[256];
    const int tid = threadIdx.x;
    float s = 0.f;
    for (int i = tid; i < n; i += 256) s += partial[i];
    red[tid] = s; __syncthreads();
    for (int o = 128; o; o >>= 1) { if (tid < o) red[tid] += red[tid + o]; __syncthreads(); }
    if (tid == 0) out[0] = -red[0] * scale;        // -mean
}

// ---------------------------------------------------------------------------
// Host driver
// ---------------------------------------------------------------------------
extern "C" void kernel_launch(void* const* d_in, const int* in_sizes, int n_in,
                              void* d_out, int out_size, void* d_ws, size_t ws_size,
                              hipStream_t stream) {
    (void)in_sizes; (void)n_in; (void)out_size; (void)ws_size;
    const int N = NROWS, D = DCOLS;
    const float* fea1 = (const float*)d_in[0];
    const float* fea2 = (const float*)d_in[1];

    // Workspace layout (~92 MB). Qf (f32) aliases V: Q(f32) is only needed
    // before the FISTA loop, V only inside it.
    char* ws = (char*)d_ws;
    size_t off = 0;
    float*  nf1 = (float*)(ws + off);  off += (size_t)N * D * 4;   // 4 MB
    float*  nf2 = (float*)(ws + off);  off += (size_t)N * D * 4;   // 4 MB
    __bf16* a1h = (__bf16*)(ws + off); off += (size_t)N * D * 2;   // 2 MB
    __bf16* a2h = (__bf16*)(ws + off); off += (size_t)N * D * 2;   // 2 MB
    __bf16* qh  = (__bf16*)(ws + off); off += (size_t)N * N * 2;   // 8 MB
    float*  Cf  = (float*)(ws + off);  off += (size_t)N * N * 4;   // 16 MB
    float*  Xf  = (float*)(ws + off);  off += (size_t)N * N * 4;   // 16 MB
    float*  Zf  = (float*)(ws + off);  off += (size_t)N * N * 4;   // 16 MB
    __bf16* zh  = (__bf16*)(ws + off); off += (size_t)N * N * 2;   // 8 MB
    float*  Vf  = (float*)(ws + off);  off += (size_t)N * N * 4;   // 16 MB (aliased Q f32)
    float*  Qf  = Vf;
    float*  v0  = (float*)(ws + off);  off += (size_t)N * 4;
    float*  v1  = (float*)(ws + off);  off += (size_t)N * 4;
    float*  part= (float*)(ws + off);  off += 4096 * 4;
    float*  scal= (float*)(ws + off);  off += 256;

    // 1) Normalize rows (f32 + bf16 copies)
    row_normalize<<<N, 256, 0, stream>>>(fea1, nf1, a1h, D);
    row_normalize<<<N, 256, 0, stream>>>(fea2, nf2, a2h, D);

    // 2) Q = A2·A2^T (f32 into Qf), bf16 copy; C = A1·A2^T
    const dim3 gQ(N / 128, N / 128);
    wmma_gemm_abt<0><<<gQ, 256, 0, stream>>>(a2h, a2h, N, N, D, Qf, nullptr, nullptr, nullptr);
    f32_to_bf16<<<(int)(((size_t)N * N + 255) / 256), 256, 0, stream>>>(Qf, qh, (size_t)N * N);
    wmma_gemm_abt<0><<<gQ, 256, 0, stream>>>(a1h, a2h, N, N, D, Cf, nullptr, nullptr, nullptr);

    // 3) Power iteration on Qf -> step = 1/(2*lmax + 1e-12) (device scalar)
    fill_f32<<<(N + 255) / 256, 256, 0, stream>>>(v0, 1.0f / N, N);
    for (int i = 0; i < POWER_ITERS; ++i) {
        matvec2048<<<N / 8, 256, 0, stream>>>(Qf, v0, v1, N);
        vec_normalize<<<1, 256, 0, stream>>>(v1, v0, N);
    }
    matvec2048<<<N / 8, 256, 0, stream>>>(Qf, v0, v1, N);
    dot_step<<<1, 256, 0, stream>>>(v0, v1, scal, N);

    // 4) FISTA: X = Z = 1/N, then 200 x (fused WMMA gradient step + projection)
    init_xz<<<(int)(((size_t)N * N + 255) / 256), 256, 0, stream>>>(Xf, Zf, zh, 1.0f / N, (size_t)N * N);
    double t = 1.0;
    for (int it = 0; it < N_ITERS; ++it) {
        wmma_gemm_abt<1><<<gQ, 256, 0, stream>>>(zh, qh, N, N, N, Vf, Zf, Cf, scal);
        const double tn = 0.5 * (1.0 + sqrt(1.0 + 4.0 * t * t));
        const float mom = (float)((t - 1.0) / tn);
        t = tn;
        fista_project<<<N, 256, 0, stream>>>(Vf, Xf, Zf, zh, N, mom);
    }

    // 5) distance = -mean((X @ nfea2 - nfea1)^2), deterministic two-pass
    dist_partial<<<dim3(D / 16, N / 16), 256, 0, stream>>>(Xf, nf2, nf1, part, N, D);
    final_reduce<<<1, 256, 0, stream>>>(part, (float*)d_out, (D / 16) * (N / 16), 1.0f / ((float)N * (float)D));
}
